// N4Net_36618891165878
// MI455X (gfx1250) — matrix operand order repaired and checked by reading, not compiled, not run on hardware
//
#include <hip/hip_runtime.h>
#include <hip/hip_bf16.h>

// CDNA5 / gfx1250 SNN forward, v3:
//  - all 58 weight B-fragments pinned in VGPRs (uses gfx1250 s_set_vgpr_msb path)
//  - LIF decay folded into post-spike update: p = beta*m - s, m' = WMMA(spk,W,p)
//  - per-wave double-buffered x tile via global_load_async_to_lds_b128
//    (ASYNCcnt-tracked DMA; hides the only remaining global-load stall)
typedef __attribute__((ext_vector_type(16))) _Float16 v16h;
typedef __attribute__((ext_vector_type(8)))  _Float16 v8h;
typedef __attribute__((ext_vector_type(8)))  float    v8f;

#define TSTEPS 20
#define WAVES  4
#define BETA   0.9f

// LDS element counts
#define W1H (128 * 32)    // w1 padded K 16->32 (halves)
#define W2H (128 * 128)
#define W3H (64 * 128)
#define W4H (16 * 64)     // w4 padded N 10->16
#define WTOT (W1H + W2H + W3H + W4H)   // 29696 halves
#define STG_HALVES (16 * 128)          // per-wave spike staging tile
#define OST_FLOATS 160                 // per-wave 16x10 f32 output staging
#define XBUF_FLOATS 256                // one 16x16 f32 x tile (1 KB)

union AFrag { v16h v; v8h h[2]; };

__device__ __forceinline__ v8f wmma16(v16h a, v16h b, v8f c) {
  return __builtin_amdgcn_wmma_f32_16x16x32_f16(false, a, false, b, (short)0, c,
                                                false, false);
}

// A fragment (16x32 f16) from LDS row-major [16][K] at column base k0.
__device__ __forceinline__ v16h load_a_lds(const _Float16* base, int K, int k0,
                                           int lane) {
  const int m = lane & 15, hi = lane >> 4;
  const _Float16* p = base + m * K + k0 + hi * 8;
  AFrag u;
  u.h[0] = *(const v8h*)p;
  u.h[1] = *(const v8h*)(p + 16);
  return u.v;
}

// B fragment (32x16 f16) from LDS weights stored (out,in) row-major [N][K].
__device__ __forceinline__ v16h load_b_lds(const _Float16* w, int K, int n0,
                                           int k0, int lane) {
  const int n = n0 + (lane & 15), hi = lane >> 4;
  const _Float16* p = w + n * K + k0 + hi * 8;
  AFrag u;
  u.h[0] = *(const v8h*)p;
  u.h[1] = *(const v8h*)(p + 16);
  return u.v;
}

// After m_new = WMMA(...): emit spike s = H(m-1) to LDS (f16, row-major [16][K])
// and return the folded next-step C operand p = beta*m - s.
__device__ __forceinline__ v8f spike_decay_store(v8f m, _Float16* stg, int K,
                                                 int n0, int lane) {
  const int col = n0 + (lane & 15), hi = lane >> 4;
  v8f p;
  #pragma unroll
  for (int r = 0; r < 8; ++r) {
    const float s = (m[r] > 1.0f) ? 1.0f : 0.0f;
    stg[(r + hi * 8) * K + col] = (_Float16)s;
    p[r] = BETA * m[r] - s;
  }
  return p;
}

// Layer-1 A fragment from the LDS x buffer (f32 -> f16, K=16 padded to 32).
__device__ __forceinline__ v16h load_a_x_lds(const float* xb, int lane) {
  const int m = lane & 15, hi = lane >> 4;
  const float* p = xb + m * 16 + hi * 8;
  AFrag u;
  #pragma unroll
  for (int i = 0; i < 8; ++i) u.h[0][i] = (_Float16)p[i];
  #pragma unroll
  for (int i = 0; i < 8; ++i) u.h[1][i] = (_Float16)0.0f;
  return u.v;
}

// Async DMA of one 1 KB x tile (16 rows x 16 f32) global -> LDS.
// Two b128 ops: lane l copies 16B at byte offset l*16 and 512 + l*16.
__device__ __forceinline__ void async_x_load(const float* xg, unsigned ldsOff,
                                             int lane) {
  unsigned long long g0 = (unsigned long long)(uintptr_t)xg + (unsigned)(lane * 16);
  unsigned l0 = ldsOff + (unsigned)(lane * 16);
  asm volatile("global_load_async_to_lds_b128 %0, %1, off"
               :: "v"(l0), "v"(g0) : "memory");
  unsigned long long g1 = g0 + 512u;
  unsigned l1 = l0 + 512u;
  asm volatile("global_load_async_to_lds_b128 %0, %1, off"
               :: "v"(l1), "v"(g1) : "memory");
}

extern __shared__ __align__(16) char smem[];

__global__ __launch_bounds__(WAVES * 32, 1)
void snn_lif_wmma_kernel(const float* __restrict__ x,
                         const float* __restrict__ w1,
                         const float* __restrict__ w2,
                         const float* __restrict__ w3,
                         const float* __restrict__ w4,
                         float* __restrict__ out, int Bsz) {
  _Float16* w1h = (_Float16*)smem;
  _Float16* w2h = w1h + W1H;
  _Float16* w3h = w2h + W2H;
  _Float16* w4h = w3h + W3H;
  _Float16* stgAll = w4h + W4H;
  float* ostAll = (float*)(stgAll + WAVES * STG_HALVES);
  float* xbufAll = ostAll + WAVES * OST_FLOATS;

  // ---- cooperative weight f32->f16 pack into LDS (B-fragment-native layout)
  const int tid = threadIdx.x;
  for (int i = tid; i < W1H; i += WAVES * 32) {
    int n = i >> 5, k = i & 31;
    w1h[i] = (k < 16) ? (_Float16)w1[n * 16 + k] : (_Float16)0.0f;
  }
  for (int i = tid; i < W2H; i += WAVES * 32) w2h[i] = (_Float16)w2[i];
  for (int i = tid; i < W3H; i += WAVES * 32) w3h[i] = (_Float16)w3[i];
  for (int i = tid; i < W4H; i += WAVES * 32) {
    int n = i >> 6, k = i & 63;
    w4h[i] = (n < 10) ? (_Float16)w4[n * 64 + k] : (_Float16)0.0f;
  }
  __syncthreads();

  const int lane = tid & 31;
  const int wave = tid >> 5;
  const int tileBase = (blockIdx.x * WAVES + wave) * 16;
  _Float16* stg = stgAll + wave * STG_HALVES;
  float* ost = ostAll + wave * OST_FLOATS;
  float* xb = xbufAll + wave * (2 * XBUF_FLOATS);
  const unsigned xbOff = (unsigned)((const char*)xb - (const char*)smem);

  // ---- pin ALL weight B-fragments in VGPRs (58 frags x 8 VGPRs = 464 VGPRs)
  v16h b1r[8], b2r[4][8], b3r[4][4], b4r[2];
  #pragma unroll
  for (int j = 0; j < 8; ++j) b1r[j] = load_b_lds(w1h, 32, j * 16, 0, lane);
  #pragma unroll
  for (int k = 0; k < 4; ++k)
    #pragma unroll
    for (int j = 0; j < 8; ++j)
      b2r[k][j] = load_b_lds(w2h, 128, j * 16, k * 32, lane);
  #pragma unroll
  for (int k = 0; k < 4; ++k)
    #pragma unroll
    for (int j = 0; j < 4; ++j)
      b3r[k][j] = load_b_lds(w3h, 128, j * 16, k * 32, lane);
  #pragma unroll
  for (int k = 0; k < 2; ++k) b4r[k] = load_b_lds(w4h, 64, 0, k * 32, lane);

  // ---- folded membrane state p = beta*m - s  (C-fragment layout, 168 VGPRs)
  v8f p1[8], p2[8], p3[4], p4;
  #pragma unroll
  for (int j = 0; j < 8; ++j)
    #pragma unroll
    for (int i = 0; i < 8; ++i) { p1[j][i] = 0.0f; p2[j][i] = 0.0f; }
  #pragma unroll
  for (int j = 0; j < 4; ++j)
    #pragma unroll
    for (int i = 0; i < 8; ++i) p3[j][i] = 0.0f;
  #pragma unroll
  for (int i = 0; i < 8; ++i) p4[i] = 0.0f;

  // ---- prologue: DMA x tile for t=0 into buffer 0
  async_x_load(x + (size_t)tileBase * 16, xbOff, lane);

  for (int t = 0; t < TSTEPS; ++t) {
    const int cur = t & 1, nxt = cur ^ 1;
    // issue DMA for the next step's tile (clamped repeat on the last step)
    const int tn = (t + 1 < TSTEPS) ? (t + 1) : t;
    async_x_load(x + ((size_t)tn * Bsz + tileBase) * 16,
                 xbOff + (unsigned)(nxt * XBUF_FLOATS * 4), lane);
    if (t + 2 < TSTEPS)  // warm L2/WGP$ two steps ahead for the DMA source
      __builtin_prefetch(x + ((size_t)(t + 2) * Bsz + tileBase) * 16 + lane * 8,
                         0, 3);
    // wait for THIS step's tile (2 newer async ops may remain outstanding;
    // async loads complete in order)
    asm volatile("s_wait_asynccnt 0x2" ::: "memory");

    // ---- layer 1: m1 = WMMA(x, w1, p1); spike+decay  (K=16 padded to 32)
    v16h a1 = load_a_x_lds(xb + cur * XBUF_FLOATS, lane);
    #pragma unroll
    for (int j = 0; j < 8; ++j) {
      v8f m = wmma16(a1, b1r[j], p1[j]);
      p1[j] = spike_decay_store(m, stg, 128, j * 16, lane);
    }

    // ---- layer 2: 128 -> 128, K = 4 x 32 (8 independent accumulator chains)
    #pragma unroll
    for (int k = 0; k < 4; ++k) {
      v16h a = load_a_lds(stg, 128, k * 32, lane);
      #pragma unroll
      for (int j = 0; j < 8; ++j) p2[j] = wmma16(a, b2r[k][j], p2[j]);
    }
    #pragma unroll
    for (int j = 0; j < 8; ++j)
      p2[j] = spike_decay_store(p2[j], stg, 128, j * 16, lane);

    // ---- layer 3: 128 -> 64
    #pragma unroll
    for (int k = 0; k < 4; ++k) {
      v16h a = load_a_lds(stg, 128, k * 32, lane);
      #pragma unroll
      for (int j = 0; j < 4; ++j) p3[j] = wmma16(a, b3r[k][j], p3[j]);
    }
    #pragma unroll
    for (int j = 0; j < 4; ++j)
      p3[j] = spike_decay_store(p3[j], stg, 64, j * 16, lane);

    // ---- layer 4: 64 -> 10 (one padded N tile), K = 2 x 32
    #pragma unroll
    for (int k = 0; k < 2; ++k) {
      v16h a = load_a_lds(stg, 64, k * 32, lane);
      p4 = wmma16(a, b4r[k], p4);
    }

    // ---- emit spk4 (16x10 f32) via LDS for coalesced global stores
    {
      const int col = lane & 15, hi = lane >> 4;
      #pragma unroll
      for (int r = 0; r < 8; ++r) {
        const float s = (p4[r] > 1.0f) ? 1.0f : 0.0f;
        if (col < 10) ost[(r + hi * 8) * 10 + col] = s;
        p4[r] = BETA * p4[r] - s;
      }
      float* op = out + ((size_t)t * Bsz + tileBase) * 10;
      #pragma unroll
      for (int i = 0; i < 5; ++i) op[lane + i * 32] = ost[lane + i * 32];
    }
  }
}

extern "C" void kernel_launch(void* const* d_in, const int* in_sizes, int n_in,
                              void* d_out, int out_size, void* d_ws,
                              size_t ws_size, hipStream_t stream) {
  const float* x  = (const float*)d_in[0];
  const float* w1 = (const float*)d_in[1];
  const float* w2 = (const float*)d_in[2];
  const float* w3 = (const float*)d_in[3];
  const float* w4 = (const float*)d_in[4];
  float* out = (float*)d_out;

  const int Bsz = in_sizes[0] / (TSTEPS * 16);   // 65536
  const int tiles = Bsz / 16;                    // 4096 waves
  const int blocks = tiles / WAVES;              // 1024 workgroups

  const size_t shmem = (size_t)(WTOT + WAVES * STG_HALVES) * sizeof(_Float16) +
                       (size_t)WAVES * OST_FLOATS * sizeof(float) +
                       (size_t)WAVES * 2 * XBUF_FLOATS * sizeof(float); // ~86 KB

  snn_lif_wmma_kernel<<<blocks, WAVES * 32, shmem, stream>>>(x, w1, w2, w3, w4,
                                                             out, Bsz);
}
